// MIKT4LPR_40553081209299
// MI455X (gfx1250) — compile-verified, matrix operand order checked by reading, not run on hardware
//
#include <hip/hip_runtime.h>
#include <hip/hip_bf16.h>

typedef __attribute__((ext_vector_type(16))) _Float16 v16h;
typedef __attribute__((ext_vector_type(8)))  float    v8f;
typedef __attribute__((__vector_size__(4 * sizeof(int)))) int v4i_vs;

#define DEV_INLINE __device__ __forceinline__
#define AS1 __attribute__((address_space(1)))
#define AS3 __attribute__((address_space(3)))

// Model dims
#define NB   32
#define LSEQ 100
#define TN   99
#define NQ   10000
#define CE   128      // C == E == 128

// Packed f16 fragment buffer sizes (in halfs)
#define PK_R1_ELEMS   (128 * 64)    // concept_readout1
#define PK_SF1_ELEMS  (384 * 128)   // skill_forget1
#define PK_SF2_ELEMS  (128 * 128)   // skill_forget2

struct KParams {
  const int*   qseq;    // (32,100)
  const int*   cseq;    // (32,100)
  const float* Qtab;    // (10000,128)
  const float* Ec;      // embed_concept (128,128)
  const float* Eq;      // embed_question (10000,128)
  const float* Ediff;   // embed_question_diff (10000,1)
  const float* Ecorr;   // embed_correctness (2,128)
  const float* Etime;   // embed_time (128,128)
  const float* Ecoarse; // embed_coarse_state (1,128)
  const float* Efine;   // embed_fine_state (128,128)
  const float *Wpro,*bpro, *Wskill,*bskill, *Wch,*bch, *Wob,*bob,
              *Wpa,*bpa, *Wqm,*bqm, *Wr1,*br1, *Wr2,*br2,
              *Wf1,*bf1, *Wf2,*bf2, *Wn1,*bn1, *Wn2,*bn2,
              *Wa1,*ba1, *Wa2,*ba2, *Wsf1,*bsf1, *Wsf2,*bsf2;
  float* qskill;  // ws: (128,128)
  float* qemb;    // ws: (10000,128)
  float* qdiff;   // ws: (10000)
  _Float16* pkWr1;   // ws: fragment-packed f16 weights
  _Float16* pkWsf1;
  _Float16* pkWsf2;
  float* out;     // (32,99)
};

DEV_INLINE float sigm(float x) { return 1.0f / (1.0f + __expf(-x)); }

// Async global->LDS 16B copy (CDNA5 GLOBAL_LOAD_ASYNC_TO_LDS_B128, ASYNCcnt).
DEV_INLINE void async_g2l_b128(const float* g, float* l) {
#if __has_builtin(__builtin_amdgcn_global_load_async_to_lds_b128)
  __builtin_amdgcn_global_load_async_to_lds_b128(
      (AS1 v4i_vs*)(unsigned long long)g,
      (AS3 v4i_vs*)(unsigned)(unsigned long long)l,
      0, 0);
#else
  l[0] = g[0]; l[1] = g[1]; l[2] = g[2]; l[3] = g[3];
#endif
}

DEV_INLINE void async_wait0() {
#if __has_builtin(__builtin_amdgcn_global_load_async_to_lds_b128)
  #if __has_builtin(__builtin_amdgcn_s_wait_asynccnt)
  __builtin_amdgcn_s_wait_asynccnt(0);
  #else
  asm volatile("s_wait_asynccnt 0x0" ::: "memory");
  #endif
#endif
}

// ---------------------------------------------------------------------------
// Fragment index math shared by packer and GEMM (CDNA5 ISA 7.12.2):
//   A/B f16 16-wide fragments: element e of lane (half = lane>>4, r = lane&15)
//   holds k = k0 + ((e<8)?e:e+8) + 8*half.  A: row M=r.  B: col N=n0+r.
//   C/D f32: element g -> M = g + 8*half, N = r.
// ---------------------------------------------------------------------------

// Wave-level GEMM: D[16 x NT] = A[16 x KT] * B[KT x NT] with B pre-packed into
// fragment order (one contiguous v16h per lane per 32x16 chunk).
template <int KT, int NT, typename AF, typename EF>
DEV_INLINE void wave_gemm_pk(int lane, AF&& A, const _Float16* Bp, EF&& epil) {
  const int half = lane >> 4;
  const int r    = lane & 15;
  const v16h* Bv = (const v16h*)Bp;
  #pragma unroll 1
  for (int n0 = 0; n0 < NT; n0 += 16) {
    v8f acc = {0.f, 0.f, 0.f, 0.f, 0.f, 0.f, 0.f, 0.f};
    #pragma unroll 1
    for (int k0 = 0; k0 < KT; k0 += 32) {
      v16h va;
      #pragma unroll
      for (int e = 0; e < 16; ++e) {
        int k = k0 + ((e < 8) ? e : e + 8) + half * 8;
        va[e] = (_Float16)A(r, k);
      }
      v16h vb = Bv[((k0 >> 5) * (NT >> 4) + (n0 >> 4)) * 32 + lane];
      acc = __builtin_amdgcn_wmma_f32_16x16x32_f16(false, va, false, vb,
                                                   (short)0, acc, false, false);
    }
    #pragma unroll
    for (int g = 0; g < 8; ++g) epil(g + 8 * half, n0 + r, acc[g]);
  }
}

// Block softmax over 128 entries in lg[] (masked entries hold -1e9).
DEV_INLINE void softmax128(float* lg, float* red, int tid) {
  if (tid < 128) red[tid] = lg[tid];
  __syncthreads();
  for (int s = 64; s > 0; s >>= 1) {
    if (tid < s) red[tid] = fmaxf(red[tid], red[tid + s]);
    __syncthreads();
  }
  float mx = red[0];
  __syncthreads();
  if (tid < 128) { float e = __expf(lg[tid] - mx); lg[tid] = e; red[tid] = e; }
  __syncthreads();
  for (int s = 64; s > 0; s >>= 1) {
    if (tid < s) red[tid] += red[tid + s];
    __syncthreads();
  }
  float sm = red[0];
  __syncthreads();
  if (tid < 128) lg[tid] /= sm;
  __syncthreads();
}

// ---------------------------------------------------------------------------
// Kernel 0a: q_skill = Ec @ Wskill + bskill   (128x128)
// ---------------------------------------------------------------------------
__global__ void mikt_qskill(KParams p) {
  int idx = blockIdx.x * blockDim.x + threadIdx.x;
  if (idx >= CE * CE) return;
  int c = idx >> 7, e = idx & 127;
  float s = p.bskill[e];
  for (int k = 0; k < CE; ++k) s += p.Ec[c * CE + k] * p.Wskill[k * CE + e];
  p.qskill[c * CE + e] = s;
}

// ---------------------------------------------------------------------------
// Kernel 0b: pre-swizzle the three hot weight matrices into f16 fragment order.
// ---------------------------------------------------------------------------
__global__ void mikt_pack(KParams p) {
  int idx = blockIdx.x * blockDim.x + threadIdx.x;
  const float* W; _Float16* dst; int NT, rel;
  if (idx < PK_R1_ELEMS)                              { W = p.Wr1;  dst = p.pkWr1;  NT = 64;  rel = idx; }
  else if (idx < PK_R1_ELEMS + PK_SF1_ELEMS)          { W = p.Wsf1; dst = p.pkWsf1; NT = 128; rel = idx - PK_R1_ELEMS; }
  else if (idx < PK_R1_ELEMS + PK_SF1_ELEMS + PK_SF2_ELEMS)
                                                      { W = p.Wsf2; dst = p.pkWsf2; NT = 128; rel = idx - PK_R1_ELEMS - PK_SF1_ELEMS; }
  else return;
  int chunk = rel >> 9, within = rel & 511;
  int lane = within >> 4, e = within & 15;
  int half = lane >> 4, r = lane & 15;
  int nchunks = NT >> 4;
  int kc = chunk / nchunks, nc = chunk % nchunks;
  int k = kc * 32 + ((e < 8) ? e : e + 8) + half * 8;
  int n = nc * 16 + r;
  dst[rel] = (_Float16)W[k * NT + n];
}

// ---------------------------------------------------------------------------
// Kernel 1: per-question embedding (one 128-thread block per question)
// ---------------------------------------------------------------------------
__global__ void mikt_qemb(KParams p) {
  __shared__ float qrow[128], qpro[128], cm[128], lg[128], red[128];
  const int q = blockIdx.x;
  const int tid = threadIdx.x;  // 128 threads

  qrow[tid] = p.Qtab[q * CE + tid];
  __syncthreads();
  red[tid] = qrow[tid];
  __syncthreads();
  for (int s = 64; s > 0; s >>= 1) {
    if (tid < s) red[tid] += red[tid + s];
    __syncthreads();
  }
  float rowsum = red[0];
  __syncthreads();

  {
    float a = 0.f;
    for (int c = 0; c < CE; ++c) a += qrow[c] * p.Ec[c * CE + tid];
    cm[tid] = a / (rowsum + 1e-8f);
  }
  {
    float a = p.bpro[tid];
    for (int k = 0; k < CE; ++k) a += p.Eq[q * CE + k] * p.Wpro[k * CE + tid];
    qpro[tid] = a;
  }
  __syncthreads();

  {
    float l;
    if (qrow[tid] == 0.f) l = -1e9f;
    else {
      float s = 0.f;
      for (int e = 0; e < CE; ++e) s += qpro[e] * p.qskill[tid * CE + e];
      l = s * 0.088388347648318447f;  // 1/sqrt(128)
    }
    lg[tid] = l;
  }
  __syncthreads();
  softmax128(lg, red, tid);

  float qd = sigm(p.Ediff[q]);
  {
    float o = 0.f, ch = p.bch[tid];
    for (int c = 0; c < CE; ++c) o += lg[c] * p.Ec[c * CE + tid];
    for (int k = 0; k < CE; ++k) ch += cm[k] * p.Wch[k * CE + tid];
    p.qemb[q * CE + tid] = o + qd * ch;
  }
  if (tid == 0) p.qdiff[q] = qd;
}

// ---------------------------------------------------------------------------
// Kernel 2: sequential scan. One workgroup per batch element; fine state lives
// in LDS for all 99 steps; row-batched MLPs on WMMA with packed-f16 B operands.
// ---------------------------------------------------------------------------
__global__ void mikt_scan(KParams p) {
  extern __shared__ float smem[];
  float* fine = smem;            // 128*128
  float* H    = smem + 16384;    // 128*128 GEMM scratch
  float* ctgE = smem + 32768;    // 128*128 gathered time embeddings
  float* base = smem + 49152;
  float* qemb  = base + 0 * 128;
  float* inter = base + 1 * 128;
  float* q2c   = base + 2 * 128;
  float* curc  = base + 3 * 128;
  float* coarse= base + 4 * 128;
  float* tgemb = base + 5 * 128;
  float* mvec  = base + 6 * 128;
  float* hbuf  = base + 7 * 128;
  float* nn    = base + 8 * 128;
  float* attnv = base + 9 * 128;
  float* tog   = base + 10 * 128;
  float* lg    = base + 11 * 128;
  float* red   = base + 12 * 128;
  int*   lastt = (int*)(base + 13 * 128);
  float* bsf1s = base + 14 * 128;  // staged small vectors
  float* bsf2s = base + 15 * 128;
  float* br1s  = base + 16 * 128;  // 64 used
  float* wr2s  = br1s + 64;        // 64 used

  const int b    = blockIdx.x;
  const int tid  = threadIdx.x;   // 256 threads = 8 wave32
  const int lane = tid & 31;
  const int m0   = (tid >> 5) * 16;  // this wave's 16-row tile

  // Warm L2 with the packed hot weights (reused 99x by every workgroup).
  if (tid < 96)  __builtin_prefetch(p.pkWsf1 + (size_t)tid * 512, 0, 1);
  if (tid < 32)  __builtin_prefetch(p.pkWsf2 + (size_t)tid * 512, 0, 1);

  for (int i = tid; i < CE * CE; i += 256) fine[i] = p.Efine[i];
  if (tid < 128) {
    coarse[tid] = p.Ecoarse[tid];
    tgemb[tid]  = p.Etime[1 * CE + tid];  // embed_time[1]
    lastt[tid]  = 0;
    bsf1s[tid]  = p.bsf1[tid];
    bsf2s[tid]  = p.bsf2[tid];
  }
  if (tid < 64) { br1s[tid] = p.br1[tid]; wr2s[tid] = p.Wr2[tid]; }
  __syncthreads();

  for (int t = 0; t < TN; ++t) {
    const int nq = p.qseq[b * LSEQ + t + 1];
    const int nc = p.cseq[b * LSEQ + t + 1];
    if (tid < 128) {
      float qe = p.qemb[nq * CE + tid];
      qemb[tid]  = qe;
      inter[tid] = qe + p.Ecorr[nc * CE + tid];
      q2c[tid]   = p.Qtab[nq * CE + tid];
    }
    __syncthreads();

    // Stage ctg_emb rows (uses last_t from previous step) via the CDNA5 async
    // global->LDS DMA path (ASYNCcnt) so the copy overlaps GEMM1 + coarse MLP.
    for (int sgm = tid; sgm < 4096; sgm += 256) {      // 4096 x 16B segments
      int c = sgm >> 5, e4 = (sgm & 31) << 2;
      int ctg = t - ((q2c[c] != 0.f) ? lastt[c] : 0);
      async_g2l_b128(p.Etime + ctg * CE + e4, ctgE + c * CE + e4);
    }

    // --- WMMA GEMM 1: H(128x64) = relu(fine @ Wr1 + br1)  (concept_readout1)
    wave_gemm_pk<128, 64>(lane,
      [&](int m, int k) { return fine[(m0 + m) * CE + k]; },
      p.pkWr1,
      [&](int m, int n, float v) { H[(m0 + m) * 64 + n] = fmaxf(v + br1s[n], 0.f); });
    __syncthreads();

    // p_c + mastery scalar
    if (tid < 128) {
      float s = p.br2[0];
      for (int k = 0; k < 64; ++k) s += H[tid * 64 + k] * wr2s[k];
      red[tid] = q2c[tid] * sigm(s);
    }
    __syncthreads();
    for (int s = 64; s > 0; s >>= 1) {
      if (tid < s) red[tid] += red[tid + s];
      __syncthreads();
    }
    float ms = red[0];
    __syncthreads();
    if (tid < 128) mvec[tid] = ms * p.Wqm[tid] + p.bqm[tid];

    // cur_coarse: coarse * sigmoid(W2 @ relu(W1 @ [coarse,tg_emb]))
    if (tid < 128) {
      float s = p.bf1[tid];
      for (int k = 0; k < 128; ++k) s += coarse[k] * p.Wf1[k * CE + tid];
      for (int k = 0; k < 128; ++k) s += tgemb[k] * p.Wf1[(128 + k) * CE + tid];
      hbuf[tid] = fmaxf(s, 0.f);
    }
    __syncthreads();
    if (tid < 128) {
      float s = p.bf2[tid];
      for (int k = 0; k < 128; ++k) s += hbuf[k] * p.Wf2[k * CE + tid];
      curc[tid] = coarse[tid] * sigm(s);
    }

    // ctgE must be resident before the next GEMM reads it.
    async_wait0();
    __syncthreads();

    // --- WMMA GEMM 2: H(128x128) = relu([fine|ctg_emb|cur_coarse] @ Wsf1 + b)
    wave_gemm_pk<384, 128>(lane,
      [&](int m, int k) {
        int row = m0 + m;
        return (k < 128) ? fine[row * CE + k]
             : (k < 256) ? ctgE[row * CE + (k - 128)]
                         : curc[k - 256];
      },
      p.pkWsf1,
      [&](int m, int n, float v) { H[(m0 + m) * CE + n] = fmaxf(v + bsf1s[n], 0.f); });
    __syncthreads();

    // --- WMMA GEMM 3: fine *= cf,  cf = q2c ? sigmoid(H @ Wsf2 + b) : 1
    wave_gemm_pk<128, 128>(lane,
      [&](int m, int k) { return H[(m0 + m) * CE + k]; },
      p.pkWsf2,
      [&](int m, int n, float v) {
        int row = m0 + m;
        float cf = (q2c[row] == 0.f) ? 1.f : sigm(v + bsf2s[n]);
        fine[row * CE + n] *= cf;
      });
    __syncthreads();

    // logits over concepts -> a1 -> now_need
    if (tid < 128) {
      float l;
      if (q2c[tid] == 0.f) l = -1e9f;
      else {
        float s = 0.f;
        for (int e = 0; e < CE; ++e) s += qemb[e] * fine[tid * CE + e];
        l = s * (1.f / 128.f);
      }
      lg[tid] = l;
    }
    __syncthreads();
    softmax128(lg, red, tid);
    if (tid < 128) {
      float s = 0.f;
      for (int c = 0; c < CE; ++c) s += lg[c] * fine[c * CE + tid];
      nn[tid] = s;
    }
    __syncthreads();

    // predict_attn
    if (tid < 128) {
      float s = p.bpa[tid];
      for (int k = 0; k < 128; ++k) s += nn[k]   * p.Wpa[k * CE + tid];
      for (int k = 0; k < 128; ++k) s += curc[k] * p.Wpa[(128 + k) * CE + tid];
      for (int k = 0; k < 128; ++k) s += qemb[k] * p.Wpa[(256 + k) * CE + tid];
      attnv[tid] = sigm(s);
    }
    __syncthreads();

    // ability MLP (K=512 -> 128 -> 1)
    if (tid < 128) {
      float s = p.ba1[tid];
      for (int k = 0; k < 128; ++k) s += ((1.f - attnv[k]) * nn[k])  * p.Wa1[k * CE + tid];
      for (int k = 0; k < 128; ++k) s += (attnv[k] * curc[k])        * p.Wa1[(128 + k) * CE + tid];
      for (int k = 0; k < 128; ++k) s += qemb[k]                     * p.Wa1[(256 + k) * CE + tid];
      for (int k = 0; k < 128; ++k) s += mvec[k]                     * p.Wa1[(384 + k) * CE + tid];
      red[tid] = fmaxf(s, 0.f) * p.Wa2[tid];
    }
    __syncthreads();
    for (int s = 64; s > 0; s >>= 1) {
      if (tid < s) red[tid] += red[tid + s];
      __syncthreads();
    }
    if (tid == 0) {
      float ability = sigm(red[0] + p.ba2[0]);
      p.out[b * TN + t] = sigm(5.f * (ability - p.qdiff[nq]));
    }
    if (tid < 128 && q2c[tid] != 0.f) lastt[tid] = t;
    __syncthreads();

    // coarse update and to_get stage 1 (both read inter only)
    if (tid < 128) {
      float so = p.bob[tid], s1 = p.bn1[tid];
      for (int k = 0; k < 128; ++k) {
        float iv = inter[k];
        so += iv * p.Wob[k * CE + tid];
        s1 += iv * p.Wn1[k * CE + tid];
      }
      coarse[tid] = curc[tid] + tanhf(so);
      hbuf[tid]   = tanhf(s1);
    }
    __syncthreads();
    if (tid < 128) {
      float s = p.bn2[tid];
      for (int k = 0; k < 128; ++k) s += hbuf[k] * p.Wn2[k * CE + tid];
      tog[tid] = tanhf(tanhf(s));
    }
    __syncthreads();

    // logits2 -> a2 -> fine += a2 (x) to_get
    if (tid < 128) {
      float l;
      if (q2c[tid] == 0.f) l = -1e9f;
      else {
        float s = 0.f;
        for (int e = 0; e < CE; ++e) s += tog[e] * fine[tid * CE + e];
        l = s * (1.f / 128.f);
      }
      lg[tid] = l;
    }
    __syncthreads();
    softmax128(lg, red, tid);
    for (int i = tid; i < CE * CE; i += 256) {
      int c = i >> 7, e = i & 127;
      fine[i] += lg[c] * tog[e];
    }
    __syncthreads();
  }
}

// ---------------------------------------------------------------------------
extern "C" void kernel_launch(void* const* d_in, const int* in_sizes, int n_in,
                              void* d_out, int out_size, void* d_ws, size_t ws_size,
                              hipStream_t stream) {
  (void)in_sizes; (void)n_in; (void)out_size; (void)ws_size;
  KParams p;
  int i = 0;
  p.qseq    = (const int*)d_in[i++];
  p.cseq    = (const int*)d_in[i++];
  p.Qtab    = (const float*)d_in[i++];
  p.Ec      = (const float*)d_in[i++];
  p.Eq      = (const float*)d_in[i++];
  p.Ediff   = (const float*)d_in[i++];
  p.Ecorr   = (const float*)d_in[i++];
  p.Etime   = (const float*)d_in[i++];
  p.Ecoarse = (const float*)d_in[i++];
  p.Efine   = (const float*)d_in[i++];
  p.Wpro  = (const float*)d_in[i++]; p.bpro  = (const float*)d_in[i++];
  p.Wskill= (const float*)d_in[i++]; p.bskill= (const float*)d_in[i++];
  p.Wch   = (const float*)d_in[i++]; p.bch   = (const float*)d_in[i++];
  p.Wob   = (const float*)d_in[i++]; p.bob   = (const float*)d_in[i++];
  p.Wpa   = (const float*)d_in[i++]; p.bpa   = (const float*)d_in[i++];
  p.Wqm   = (const float*)d_in[i++]; p.bqm   = (const float*)d_in[i++];
  p.Wr1   = (const float*)d_in[i++]; p.br1   = (const float*)d_in[i++];
  p.Wr2   = (const float*)d_in[i++]; p.br2   = (const float*)d_in[i++];
  p.Wf1   = (const float*)d_in[i++]; p.bf1   = (const float*)d_in[i++];
  p.Wf2   = (const float*)d_in[i++]; p.bf2   = (const float*)d_in[i++];
  p.Wn1   = (const float*)d_in[i++]; p.bn1   = (const float*)d_in[i++];
  p.Wn2   = (const float*)d_in[i++]; p.bn2   = (const float*)d_in[i++];
  p.Wa1   = (const float*)d_in[i++]; p.ba1   = (const float*)d_in[i++];
  p.Wa2   = (const float*)d_in[i++]; p.ba2   = (const float*)d_in[i++];
  p.Wsf1  = (const float*)d_in[i++]; p.bsf1  = (const float*)d_in[i++];
  p.Wsf2  = (const float*)d_in[i++]; p.bsf2  = (const float*)d_in[i++];

  float* ws = (float*)d_ws;
  p.qskill = ws;
  p.qemb   = ws + 128 * 128;
  p.qdiff  = p.qemb + NQ * 128;
  _Float16* ph = (_Float16*)(p.qdiff + NQ);   // 32B-aligned (offset 5,225,536 B)
  p.pkWr1  = ph;
  p.pkWsf1 = ph + PK_R1_ELEMS;
  p.pkWsf2 = ph + PK_R1_ELEMS + PK_SF1_ELEMS;
  p.out    = (float*)d_out;

  mikt_qskill<<<64, 256, 0, stream>>>(p);
  mikt_pack<<<(PK_R1_ELEMS + PK_SF1_ELEMS + PK_SF2_ELEMS + 255) / 256, 256, 0, stream>>>(p);
  mikt_qemb<<<NQ, 128, 0, stream>>>(p);
  size_t smem = (size_t)(49152 + 17 * 128) * sizeof(float);  // ~205 KB of 320 KB WGP LDS
  mikt_scan<<<NB, 256, smem, stream>>>(p);
}
